// RWKV_Tmix_x070_Mose_cxa076_49426483642922
// MI455X (gfx1250) — matrix operand order repaired and checked by reading, not compiled, not run on hardware
//
#include <hip/hip_runtime.h>
#include <hip/hip_bf16.h>
#include <math.h>

typedef __bf16 bf16;
typedef __attribute__((ext_vector_type(16))) __bf16 v16bf;
typedef __attribute__((ext_vector_type(8)))  __bf16 v8bf;
typedef __attribute__((ext_vector_type(8)))  float  v8f;
typedef unsigned int ui4 __attribute__((ext_vector_type(4)));
typedef int          i8v __attribute__((ext_vector_type(8)));
typedef int          i4v __attribute__((ext_vector_type(4)));

#define B_   2
#define T_   1024
#define C_   2048
#define H_   32
#define HKV_ 8
#define N_   64
#define NREP_ 4
#define BT_  (B_ * T_)

#if __has_builtin(__builtin_amdgcn_tensor_load_to_lds) && __has_builtin(__builtin_amdgcn_s_wait_tensorcnt)
#define USE_TDM 1
#else
#define USE_TDM 0
#endif

// ---------------------------------------------------------------------------
// f32 -> bf16 converts. Weights are converted TRANSPOSED (+zero pad) once per
// launch so every GEMM B-operand is N-major: tile loads become pure 16B
// vector ops with no in-kernel transpose scatter.
// ---------------------------------------------------------------------------
__global__ void cvt_f32_bf16(const float* __restrict__ s, bf16* __restrict__ d, size_t n) {
  size_t i = (size_t)blockIdx.x * blockDim.x + threadIdx.x;
  size_t stride = (size_t)gridDim.x * blockDim.x;
  for (; i < n; i += stride) d[i] = (bf16)s[i];
}

// src: rows(K) x cols(N) row-major f32. dst: Npad x Kpad bf16, dst[n*Kpad+k].
__global__ void cvt_T_pad(const float* __restrict__ s, bf16* __restrict__ d,
                          int K, int N, int Kpad, int Npad) {
  size_t total = (size_t)Npad * Kpad;
  size_t stride = (size_t)gridDim.x * blockDim.x;
  for (size_t idx = (size_t)blockIdx.x * blockDim.x + threadIdx.x; idx < total; idx += stride) {
    int n = (int)(idx / Kpad), k = (int)(idx % Kpad);
    d[idx] = (k < K && n < N) ? (bf16)s[(size_t)k * N + n] : (bf16)0.0f;
  }
}

// ---------------------------------------------------------------------------
// WMMA bf16 GEMM: C[M,N] = A[M,K] @ B[K,N], f32 accumulate.
// B passed TRANSPOSED: Bt[n*K + k]. Block tile 128 x BN, 8 waves (4x2),
// wave tile 32 x BN/2 = 2 x FN WMMA frags, BK=64 (2 k-steps per LDS fill).
// Two-stage software pipeline: tile k+1 staged into VGPRs while WMMAs for
// tile k execute, hiding global latency under the matrix-core phase.
// All dims padded (M%128, N%64|128, K%64 == 0) -> zero guards anywhere.
// 32-bit offsets throughout (all buffers < 2^31 elems) -> saddr+voffset VMEM.
// ---------------------------------------------------------------------------
template<int BN, int ACT, bool WF32, bool WBF16>
__global__ void __launch_bounds__(256) gemm_bf16_wmma(
    const bf16* __restrict__ A, const bf16* __restrict__ Bt,
    float* __restrict__ Cf, bf16* __restrict__ Cb, int N, int K)
{
  constexpr int FN   = BN / 32;           // 16-col fragments per wave
  constexpr int BREP = BN / 32;           // B-tile 8-elem chunks per thread
  __shared__ bf16 ldsA[128 * 72];         // stride 72 elems = 144B (16B aligned)
  __shared__ bf16 ldsB[BN * 72];          // [n][k], same padded stride

  const int tid  = threadIdx.x;
  const int lane = tid & 31;
  const int wave = tid >> 5;
  const int wm = wave & 3, wn = wave >> 2;
  const unsigned rowBase = (unsigned)blockIdx.x * 128u;
  const unsigned colBase = (unsigned)blockIdx.y * (unsigned)BN;
  const unsigned uK = (unsigned)K;

  // per-thread tile chunk coordinates (8-elem chunks)
  int aRow[4], aCol[4];
  #pragma unroll
  for (int rep = 0; rep < 4; rep++) {
    int ch = tid + rep * 256;
    aRow[rep] = ch >> 3;
    aCol[rep] = (ch & 7) * 8;
  }
  int bRow[BREP], bCol[BREP];
  #pragma unroll
  for (int rep = 0; rep < BREP; rep++) {
    int ch = tid + rep * 256;
    bRow[rep] = ch >> 3;
    bCol[rep] = (ch & 7) * 8;
  }

  v8f acc[2][FN];
  #pragma unroll
  for (int i = 0; i < 2; i++)
    #pragma unroll
    for (int j = 0; j < FN; j++)
      acc[i][j] = v8f{0.f,0.f,0.f,0.f,0.f,0.f,0.f,0.f};

  // prologue: stage tile kb=0 into registers
  v8bf aReg[4], bReg[BREP];
  #pragma unroll
  for (int rep = 0; rep < 4; rep++)
    aReg[rep] = *(const v8bf*)(A + (rowBase + (unsigned)aRow[rep]) * uK + (unsigned)aCol[rep]);
  #pragma unroll
  for (int rep = 0; rep < BREP; rep++)
    bReg[rep] = *(const v8bf*)(Bt + (colBase + (unsigned)bRow[rep]) * uK + (unsigned)bCol[rep]);

  const int arow = lane & 15;        // A frag row in 16-tile
  const int akb  = (lane >> 4) * 8;  // A frag K base (+16 wrap)
  const int bks  = (lane >> 4) * 8;  // B frag K base (+16 wrap)

  for (unsigned kb = 0; kb < uK; kb += 64) {
    // commit staged tile to LDS
    #pragma unroll
    for (int rep = 0; rep < 4; rep++)
      *(v8bf*)(&ldsA[aRow[rep] * 72 + aCol[rep]]) = aReg[rep];
    #pragma unroll
    for (int rep = 0; rep < BREP; rep++)
      *(v8bf*)(&ldsB[bRow[rep] * 72 + bCol[rep]]) = bReg[rep];
    __syncthreads();

    // issue next tile's global loads; waited only at next iteration's commit
    const unsigned kn = kb + 64;
    if (kn < uK) {
      #pragma unroll
      for (int rep = 0; rep < 4; rep++)
        aReg[rep] = *(const v8bf*)(A + (rowBase + (unsigned)aRow[rep]) * uK + kn + (unsigned)aCol[rep]);
      #pragma unroll
      for (int rep = 0; rep < BREP; rep++)
        bReg[rep] = *(const v8bf*)(Bt + (colBase + (unsigned)bRow[rep]) * uK + kn + (unsigned)bCol[rep]);
    }

    #pragma unroll
    for (int ks = 0; ks < 64; ks += 32) {
      v16bf af[2], bfr[FN];
      #pragma unroll
      for (int fm = 0; fm < 2; fm++) {
        const bf16* pa = &ldsA[(wm * 32 + fm * 16 + arow) * 72 + ks + akb];
        v8bf lo = *(const v8bf*)pa;
        v8bf hi = *(const v8bf*)(pa + 16);
        #pragma unroll
        for (int e = 0; e < 8; e++) { af[fm][e] = lo[e]; af[fm][8 + e] = hi[e]; }
      }
      #pragma unroll
      for (int fn = 0; fn < FN; fn++) {
        const bf16* pb = &ldsB[(wn * (BN / 2) + fn * 16 + (lane & 15)) * 72 + ks + bks];
        v8bf lo = *(const v8bf*)pb;
        v8bf hi = *(const v8bf*)(pb + 16);
        #pragma unroll
        for (int e = 0; e < 8; e++) { bfr[fn][e] = lo[e]; bfr[fn][8 + e] = hi[e]; }
      }
      #pragma unroll
      for (int fm = 0; fm < 2; fm++)
        #pragma unroll
        for (int fn = 0; fn < FN; fn++)
          acc[fm][fn] = __builtin_amdgcn_wmma_f32_16x16x32_bf16(
              false, af[fm], false, bfr[fn], (short)0, acc[fm][fn], false, false);
    }
    __syncthreads();
  }

  // epilogue: VGPR l holds M = l + 8*(lane>=16), N = lane&15; 32-bit indices
  const unsigned mhi = (unsigned)((lane >> 4) * 8);
  const unsigned nn  = (unsigned)(lane & 15);
  const unsigned uN  = (unsigned)N;
  #pragma unroll
  for (int fm = 0; fm < 2; fm++)
    #pragma unroll
    for (int fn = 0; fn < FN; fn++)
      #pragma unroll
      for (int l = 0; l < 8; l++) {
        unsigned grow = rowBase + (unsigned)(wm * 32 + fm * 16) + mhi + (unsigned)l;
        unsigned gcol = colBase + (unsigned)(wn * (BN / 2) + fn * 16) + nn;
        unsigned idx  = grow * uN + gcol;
        float val = acc[fm][fn][l];
        if (ACT == 1) val = tanhf(val);
        else if (ACT == 2) val = 1.0f / (1.0f + expf(-val));
        if (WF32)  Cf[idx] = val;
        if (WBF16) Cb[idx] = (bf16)val;
      }
}

// ---------------------------------------------------------------------------
// Fused pointwise stage 1: one 64-thread (2-wave) block per (b,t,h).
// ---------------------------------------------------------------------------
__device__ __forceinline__ float reduce64(float v, float* red2, int i) {
  v += __shfl_xor(v, 1);  v += __shfl_xor(v, 2);  v += __shfl_xor(v, 4);
  v += __shfl_xor(v, 8);  v += __shfl_xor(v, 16);
  if ((i & 31) == 0) red2[i >> 5] = v;
  __syncthreads();
  float s = red2[0] + red2[1];
  __syncthreads();
  return s;
}

__device__ __forceinline__ float sigf(float x) { return 1.0f / (1.0f + expf(-x)); }

__global__ void __launch_bounds__(64) pointwise1(
    const float* __restrict__ xWr, const float* __restrict__ xWk, const float* __restrict__ xWv,
    const float* __restrict__ w0,  const float* __restrict__ w2o,
    const float* __restrict__ a0,  const float* __restrict__ a2o,
    const float* __restrict__ v0,  const float* __restrict__ v2o,
    const float* __restrict__ v_first, const float* __restrict__ g2o,
    const float* __restrict__ r_k, const float* __restrict__ r_norm_w,
    const float* __restrict__ k_norm_w, const float* __restrict__ attn,
    float* __restrict__ scanin, float* __restrict__ dotrk,
    float* __restrict__ vkeep,  float* __restrict__ gkeep)
{
  __shared__ float red2[2];
  const int i   = threadIdx.x;
  const int h   = blockIdx.x % H_;
  const int bt  = blockIdx.x / H_;
  const int hkv = h / NREP_;

  float rraw = xWr[(unsigned)bt * C_ + h * N_ + i];
  float kraw = xWk[(unsigned)bt * (HKV_ * N_) + hkv * N_ + i];
  float vraw = xWv[(unsigned)bt * (HKV_ * N_) + hkv * N_ + i];

  float rms_r = reduce64(rraw * rraw, red2, i) * (1.0f / 64.0f);
  float r  = r_norm_w[i] * rraw * rsqrtf(rms_r + 1e-6f);
  float rms_k = reduce64(kraw * kraw, red2, i) * (1.0f / 64.0f);
  float k0 = k_norm_w[i] * kraw * rsqrtf(rms_k + 1e-6f);

  const int j = h * N_ + i;
  const unsigned cidx = (unsigned)bt * C_ + j;
  float wz   = w0[j] + w2o[cidx];
  float wl   = -log1pf(expf(-wz)) - 0.5f;           // -softplus(-z) - 0.5
  float vmix = sigf(v0[j] + v2o[cidx]);
  float vf   = vraw + (v_first[cidx] - vraw) * vmix;
  float aa   = sigf(a0[j] + a2o[cidx]);
  float g    = g2o[cidx];

  float kn2 = reduce64(k0 * k0, red2, i);
  float kk  = k0 / fmaxf(sqrtf(kn2), 1e-12f);
  float kf  = k0 * (1.0f - wl + aa);
  float wdec = expf(-expf(wl));
  float m   = attn[bt];

  const int b = bt / T_, t = bt % T_;
  float* dst = scanin + ((unsigned)(b * H_ + h) * T_ + t) * 384u;
  dst[      i] = r;
  dst[ 64 + i] = wdec;
  dst[128 + i] = kf * m;
  dst[192 + i] = vf * m;
  dst[256 + i] = -kk * m;
  dst[320 + i] = kk * aa * m;
  vkeep[cidx] = vf;
  gkeep[cidx] = g;

  float dp = reduce64(r * kf * r_k[j], red2, i);
  if (i == 0) dotrk[(unsigned)bt * H_ + h] = dp;
}

// ---------------------------------------------------------------------------
// TDM helper: 1-D tensor DMA of 384 f32 (1.5KB) global -> LDS.
// D# group0: count=1, lds_addr, global_addr, type=2.
// D# group1: data_size=4B, tensor_dim0=tile_dim0=stride=384.
// ---------------------------------------------------------------------------
#if USE_TDM
__device__ __forceinline__ void tdm_load_row(const float* gsrc, void* ldst) {
  unsigned long long ga = (unsigned long long)(size_t)gsrc;
  unsigned lds_lo = (unsigned)(size_t)ldst;   // low 32 bits = LDS byte offset
  ui4 g0;
  g0[0] = 1u;                                  // count=1 (valid user descriptor)
  g0[1] = lds_lo;                              // lds_addr  [63:32]
  g0[2] = (unsigned)ga;                        // global_addr[95:64]
  g0[3] = (unsigned)(ga >> 32) | (2u << 30);   // global_addr hi | type=2
  i8v g1;
  g1[0] = 0x20000;                             // data_size=2 (4 bytes)
  g1[1] = (int)(384u << 16);                   // tensor_dim0 = 384
  g1[2] = 0;
  g1[3] = (int)(384u << 16);                   // tile_dim0 = 384
  g1[4] = 0;                                   // tile_dim1/2 unused (1-D)
  g1[5] = 384;                                 // tensor_dim0_stride
  g1[6] = 0;
  g1[7] = 0;
  i4v gz = {0, 0, 0, 0};
#if defined(__clang_major__) && (__clang_major__ >= 23)
  i8v gz8 = {0, 0, 0, 0, 0, 0, 0, 0};
  __builtin_amdgcn_tensor_load_to_lds(g0, g1, gz, gz, gz8, 0);
#else
  __builtin_amdgcn_tensor_load_to_lds(g0, g1, gz, gz, 0);
#endif
}
#endif

// ---------------------------------------------------------------------------
// WKV7 scan: one 256-thread block per (b,h). Thread owns row=tid>>2,
// cols [16*(tid&3), +16) of the 64x64 state in VGPRs; 4-lane shfl reductions.
// Per-step 1.5KB vector block staged in double-buffered LDS via TDM
// (wave 0 issues; s_wait_tensorcnt + barrier before consumption).
// ---------------------------------------------------------------------------
__global__ void __launch_bounds__(256) wkv7_scan(
    const float* __restrict__ scanin, float* __restrict__ y)
{
  const int tid = threadIdx.x;
  const int h = blockIdx.x % H_;
  const int b = blockIdx.x / H_;
  const int row = tid >> 2;
  const int q   = tid & 3;
  const int jb  = q * 16;
  __shared__ float buf[2][384];
  const float* src = scanin + (unsigned)(b * H_ + h) * (T_ * 384u);

  float S[16];
  #pragma unroll
  for (int c = 0; c < 16; c++) S[c] = 0.f;

#if USE_TDM
  const bool issuer = (tid < 32);       // wave 0 only (TDM ignores EXEC)
  if (issuer) {
    tdm_load_row(src, &buf[0][0]);
    __builtin_amdgcn_s_wait_tensorcnt(0);
  }
  __syncthreads();
#else
  buf[0][tid] = src[tid];
  if (tid < 128) buf[0][256 + tid] = src[256 + tid];
  __syncthreads();
#endif

  for (int t = 0; t < T_; t++) {
    const float* cur = buf[t & 1];
    const bool have = (t + 1 < T_);
#if USE_TDM
    if (have && issuer)
      tdm_load_row(src + (unsigned)(t + 1) * 384u, &buf[(t + 1) & 1][0]);
#else
    float pre0 = 0.f, pre1 = 0.f;
    if (have) {
      const float* ns = src + (unsigned)(t + 1) * 384u;
      pre0 = ns[tid];
      if (tid < 128) pre1 = ns[256 + tid];
      if (t + 2 < T_) __builtin_prefetch(src + (unsigned)(t + 2) * 384u + tid, 0, 1);
    }
#endif
    const float* rv = cur;
    const float* wv = cur + 64;
    const float* kv = cur + 128;
    const float* vv = cur + 192;
    const float* av = cur + 256;
    const float* bv = cur + 320;

    float sa = 0.f;
    #pragma unroll
    for (int c = 0; c < 16; c++) sa += S[c] * av[jb + c];
    sa += __shfl_xor(sa, 1);
    sa += __shfl_xor(sa, 2);

    const float vi = vv[row];
    float yp = 0.f;
    #pragma unroll
    for (int c = 0; c < 16; c++) {
      int jj = jb + c;
      S[c] = S[c] * wv[jj] + sa * bv[jj] + vi * kv[jj];
      yp += S[c] * rv[jj];
    }
    yp += __shfl_xor(yp, 1);
    yp += __shfl_xor(yp, 2);
    if (q == 0)
      y[((unsigned)(b * T_ + t) * H_ + h) * N_ + row] = yp * 0.125f;  // N^-0.5

#if USE_TDM
    if (issuer) __builtin_amdgcn_s_wait_tensorcnt(0);
#else
    if (have) {
      float* nxt = (float*)buf[(t + 1) & 1];
      nxt[tid] = pre0;
      if (tid < 128) nxt[256 + tid] = pre1;
    }
#endif
    __syncthreads();
  }
}

// ---------------------------------------------------------------------------
// Pointwise stage 2: z = (y + dotrk*v) * g  ->  bf16 input of final GEMM
// ---------------------------------------------------------------------------
__global__ void pointwise2(const float* __restrict__ y, const float* __restrict__ dotrk,
                           const float* __restrict__ vkeep, const float* __restrict__ gkeep,
                           bf16* __restrict__ zbf, unsigned n)
{
  unsigned idx = blockIdx.x * blockDim.x + threadIdx.x;
  if (idx >= n) return;
  unsigned bt = idx / C_;
  unsigned j = idx % C_;
  unsigned h = j >> 6;
  float z = (y[idx] + dotrk[bt * H_ + h] * vkeep[idx]) * gkeep[idx];
  zbf[idx] = (bf16)z;
}

__global__ void copy_f32(const float* __restrict__ s, float* __restrict__ d, size_t n) {
  size_t i = (size_t)blockIdx.x * blockDim.x + threadIdx.x;
  size_t stride = (size_t)gridDim.x * blockDim.x;
  for (; i < n; i += stride) d[i] = s[i];
}

// ---------------------------------------------------------------------------
extern "C" void kernel_launch(void* const* d_in, const int* in_sizes, int n_in,
                              void* d_out, int out_size, void* d_ws, size_t ws_size,
                              hipStream_t stream) {
  (void)in_sizes; (void)n_in; (void)out_size; (void)ws_size;
  const float* x        = (const float*)d_in[0];
  const float* v_first  = (const float*)d_in[1];
  const float* w0       = (const float*)d_in[2];
  const float* w1       = (const float*)d_in[3];
  const float* w2       = (const float*)d_in[4];
  const float* a0       = (const float*)d_in[5];
  const float* a1       = (const float*)d_in[6];
  const float* a2       = (const float*)d_in[7];
  const float* v0       = (const float*)d_in[8];
  const float* v1       = (const float*)d_in[9];
  const float* v2       = (const float*)d_in[10];
  const float* g1       = (const float*)d_in[11];
  const float* g2       = (const float*)d_in[12];
  const float* r_k      = (const float*)d_in[13];
  const float* Wr       = (const float*)d_in[14];
  const float* Wk       = (const float*)d_in[15];
  const float* Wv       = (const float*)d_in[16];
  const float* Wo       = (const float*)d_in[17];
  const float* r_norm_w = (const float*)d_in[18];
  const float* k_norm_w = (const float*)d_in[19];
  const float* attn     = (const float*)d_in[20];

  char* ws = (char*)d_ws;
  size_t off = 0;
  auto alloc = [&](size_t bytes) -> void* {
    off = (off + 255) & ~(size_t)255;
    void* p = ws + off;
    off += bytes;
    return p;
  };

  // bf16 operands; weights stored TRANSPOSED (N-major) and padded
  bf16* xbf  = (bf16*)alloc((size_t)BT_ * C_ * 2);
  bf16* wrb  = (bf16*)alloc((size_t)C_ * C_ * 2);     // [2048n][2048k]
  bf16* wkb  = (bf16*)alloc((size_t)512 * C_ * 2);    // [512n][2048k]
  bf16* wvb  = (bf16*)alloc((size_t)512 * C_ * 2);
  bf16* w1b  = (bf16*)alloc((size_t)128 * C_ * 2);    // [128n pad][2048k]
  bf16* w2b  = (bf16*)alloc((size_t)C_ * 128 * 2);    // [2048n][128k pad]
  bf16* a1b  = (bf16*)alloc((size_t)128 * C_ * 2);
  bf16* a2b  = (bf16*)alloc((size_t)C_ * 128 * 2);
  bf16* v1b  = (bf16*)alloc((size_t)64 * C_ * 2);
  bf16* v2b  = (bf16*)alloc((size_t)C_ * 64 * 2);
  bf16* g1b  = (bf16*)alloc((size_t)256 * C_ * 2);
  bf16* g2b  = (bf16*)alloc((size_t)C_ * 256 * 2);
  bf16* wob  = (bf16*)alloc((size_t)C_ * C_ * 2);
  // intermediates
  float* xWr = (float*)alloc((size_t)BT_ * C_ * 4);
  float* xWk = (float*)alloc((size_t)BT_ * 512 * 4);
  float* xWv = (float*)alloc((size_t)BT_ * 512 * 4);
  bf16*  hw  = (bf16*)alloc((size_t)BT_ * 128 * 2);
  bf16*  ha  = (bf16*)alloc((size_t)BT_ * 128 * 2);
  bf16*  hv  = (bf16*)alloc((size_t)BT_ * 64 * 2);
  bf16*  hg  = (bf16*)alloc((size_t)BT_ * 256 * 2);
  float* w2o = (float*)alloc((size_t)BT_ * C_ * 4);
  float* a2o = (float*)alloc((size_t)BT_ * C_ * 4);
  float* v2o = (float*)alloc((size_t)BT_ * C_ * 4);
  float* g2o = (float*)alloc((size_t)BT_ * C_ * 4);
  float* scanin = (float*)alloc((size_t)B_ * H_ * T_ * 384 * 4);
  float* dotrk  = (float*)alloc((size_t)BT_ * H_ * 4);
  float* vkeep  = (float*)alloc((size_t)BT_ * C_ * 4);
  float* gkeep  = (float*)alloc((size_t)BT_ * C_ * 4);
  float* ybuf   = (float*)alloc((size_t)BT_ * C_ * 4);
  bf16*  zbf    = (bf16*)alloc((size_t)BT_ * C_ * 2);

  auto cvt = [&](const float* s, bf16* d, size_t n) {
    int blocks = (int)((n + 255) / 256);
    if (blocks > 1024) blocks = 1024;
    cvt_f32_bf16<<<blocks, 256, 0, stream>>>(s, d, n);
  };
  auto cvtT = [&](const float* s, bf16* d, int K, int N, int Kpad, int Npad) {
    size_t n = (size_t)Npad * Kpad;
    int blocks = (int)((n + 255) / 256);
    if (blocks > 1024) blocks = 1024;
    cvt_T_pad<<<blocks, 256, 0, stream>>>(s, d, K, N, Kpad, Npad);
  };
  cvt (x,  xbf, (size_t)BT_ * C_);
  cvtT(Wr, wrb, C_,  C_,  C_,  C_);
  cvtT(Wk, wkb, C_,  512, C_,  512);
  cvtT(Wv, wvb, C_,  512, C_,  512);
  cvtT(w1, w1b, C_,  96,  C_,  128);
  cvtT(w2, w2b, 96,  C_,  128, C_);
  cvtT(a1, a1b, C_,  96,  C_,  128);
  cvtT(a2, a2b, 96,  C_,  128, C_);
  cvtT(v1, v1b, C_,  64,  C_,  64);
  cvtT(v2, v2b, 64,  C_,  64,  C_);
  cvtT(g1, g1b, C_,  256, C_,  256);
  cvtT(g2, g2b, 256, C_,  256, C_);
  cvtT(Wo, wob, C_,  C_,  C_,  C_);

  auto grid128 = [](int N) { return dim3(BT_ / 128, N / 128); };

  // Stage-1 projections from x
  gemm_bf16_wmma<128, 0, true,  false><<<grid128(C_),  256, 0, stream>>>(xbf, wrb, xWr, nullptr, C_,  C_);
  gemm_bf16_wmma<128, 0, true,  false><<<grid128(512), 256, 0, stream>>>(xbf, wkb, xWk, nullptr, 512, C_);
  gemm_bf16_wmma<128, 0, true,  false><<<grid128(512), 256, 0, stream>>>(xbf, wvb, xWv, nullptr, 512, C_);
  gemm_bf16_wmma<128, 1, false, true ><<<grid128(128), 256, 0, stream>>>(xbf, w1b, nullptr, hw, 128, C_); // tanh
  gemm_bf16_wmma<128, 0, false, true ><<<grid128(128), 256, 0, stream>>>(xbf, a1b, nullptr, ha, 128, C_);
  gemm_bf16_wmma< 64, 0, false, true ><<<dim3(BT_/128, 1), 256, 0, stream>>>(xbf, v1b, nullptr, hv, 64, C_);
  gemm_bf16_wmma<128, 2, false, true ><<<grid128(256), 256, 0, stream>>>(xbf, g1b, nullptr, hg, 256, C_); // sigmoid
  // Stage-2 low-rank expansions (K padded where needed)
  gemm_bf16_wmma<128, 0, true,  false><<<grid128(C_), 256, 0, stream>>>(hw, w2b, w2o, nullptr, C_, 128);
  gemm_bf16_wmma<128, 0, true,  false><<<grid128(C_), 256, 0, stream>>>(ha, a2b, a2o, nullptr, C_, 128);
  gemm_bf16_wmma<128, 0, true,  false><<<grid128(C_), 256, 0, stream>>>(hv, v2b, v2o, nullptr, C_, 64);
  gemm_bf16_wmma<128, 0, true,  false><<<grid128(C_), 256, 0, stream>>>(hg, g2b, g2o, nullptr, C_, 256);

  // Fused pointwise (norms, decay, kk, packing) — one block per (b,t,h)
  pointwise1<<<BT_ * H_, 64, 0, stream>>>(xWr, xWk, xWv, w0, w2o, a0, a2o, v0, v2o,
                                          v_first, g2o, r_k, r_norm_w, k_norm_w, attn,
                                          scanin, dotrk, vkeep, gkeep);

  // Sequential WKV7 scan — one block per (b,h), TDM-staged step vectors
  wkv7_scan<<<B_ * H_, 256, 0, stream>>>(scanin, ybuf);

  // Merge + output projection straight into d_out
  size_t nz = (size_t)BT_ * C_;
  pointwise2<<<(int)((nz + 255) / 256), 256, 0, stream>>>(ybuf, dotrk, vkeep, gkeep, zbf, (unsigned)nz);
  gemm_bf16_wmma<128, 0, true, false><<<grid128(C_), 256, 0, stream>>>(zbf, wob, (float*)d_out, nullptr, C_, C_);

  // Second output of the tuple: v_first passthrough
  copy_f32<<<1024, 256, 0, stream>>>(v_first, (float*)d_out + nz, nz);
}
